// MolecularAttentiveFP_20590073217132
// MI455X (gfx1250) — compile-verified, matrix operand order checked by reading, not run on hardware
//
#include <hip/hip_runtime.h>
#include <hip/hip_bf16.h>
#include <math.h>

typedef __attribute__((ext_vector_type(16))) _Float16 v16h;
typedef __attribute__((ext_vector_type(8)))  _Float16 v8h;
typedef __attribute__((ext_vector_type(8)))  float    v8f;

#define NEG_SLOPE 0.01f

__device__ __forceinline__ float leakyf(float v) { return v >= 0.f ? v : NEG_SLOPE * v; }

// ---------------------------------------------------------------------------
// WMMA fragment load: 16 halves for one lane of a 16x32 f16 A (or transposed-B)
// tile. Per ISA layout, lane (half = lane>>4) needs K = half*8..half*8+7 and
// K = 16+half*8..16+half*8+7 -> two contiguous 16B loads.
// ---------------------------------------------------------------------------
__device__ __forceinline__ v16h frag_load(const _Float16* p) {
  v8h lo = *(const v8h*)(p);
  v8h hi = *(const v8h*)(p + 16);
  return __builtin_shufflevector(lo, hi, 0,1,2,3,4,5,6,7,8,9,10,11,12,13,14,15);
}

// ---------------------------------------------------------------------------
// Generic GEMM: C[M,N] (f32, optional f16 mirror) = act(A[M,K]f16 @ Bt[N,K]^T + bias)
// Block: 128 threads = 4 waves; each wave owns a 32x64 output tile
// (2 A-frags x 4 B-frags = 8 WMMAs per K-step -> 1.5 b128 loads per wmma).
// M multiple of 32; N multiple of 64; K multiple of 32 (all hold here).
// act: 0 = none, 1 = leaky_relu(0.01)
// ---------------------------------------------------------------------------
__global__ __launch_bounds__(128)
void k_gemm_wmma(const _Float16* __restrict__ A, const _Float16* __restrict__ Bt,
                 const float* __restrict__ bias, float* __restrict__ C,
                 _Float16* __restrict__ C16, int M, int N, int K, int act)
{
  const int lane = threadIdx.x & 31;
  const int wave = threadIdx.x >> 5;
  const int row0 = (blockIdx.x * 4 + wave) * 32;
  const int col0 = blockIdx.y * 64;
  if (row0 >= M || col0 >= N) return;
  const int half = lane >> 4;     // 0 or 1
  const int l16  = lane & 15;

  v8f acc[2][4];
#pragma unroll
  for (int i = 0; i < 2; ++i)
#pragma unroll
    for (int j = 0; j < 4; ++j) acc[i][j] = (v8f){};

  const _Float16* ap[2];
  const _Float16* bp[4];
#pragma unroll
  for (int i = 0; i < 2; ++i) ap[i] = A  + (size_t)(row0 + i * 16 + l16) * K + half * 8;
#pragma unroll
  for (int j = 0; j < 4; ++j) bp[j] = Bt + (size_t)(col0 + j * 16 + l16) * K + half * 8;

  for (int kk = 0; kk < K; kk += 32) {
    v16h a[2], b[4];
#pragma unroll
    for (int i = 0; i < 2; ++i) a[i] = frag_load(ap[i] + kk);
#pragma unroll
    for (int j = 0; j < 4; ++j) b[j] = frag_load(bp[j] + kk);
#pragma unroll
    for (int i = 0; i < 2; ++i)
#pragma unroll
      for (int j = 0; j < 4; ++j)
        acc[i][j] = __builtin_amdgcn_wmma_f32_16x16x32_f16(false, a[i], false, b[j],
                                                           (short)0, acc[i][j], false, false);
  }

#pragma unroll
  for (int i = 0; i < 2; ++i)
#pragma unroll
    for (int j = 0; j < 4; ++j) {
      const int col = col0 + j * 16 + l16;
      float bval = bias ? bias[col] : 0.f;
#pragma unroll
      for (int v = 0; v < 8; ++v) {
        int row = row0 + i * 16 + half * 8 + v;
        float xv = acc[i][j][v] + bval;
        if (act == 1) xv = leakyf(xv);
        size_t o = (size_t)row * N + col;
        C[o] = xv;
        if (C16) C16[o] = (_Float16)xv;
      }
    }
}

// --------------------------- small utility kernels -------------------------
__global__ void k_zerof(float* p, size_t n) {
  size_t t = (size_t)blockIdx.x * blockDim.x + threadIdx.x;
  if (t < n) p[t] = 0.f;
}

__global__ void k_cast16(const float* __restrict__ s, _Float16* __restrict__ d, size_t n) {
  size_t t = (size_t)blockIdx.x * blockDim.x + threadIdx.x;
  if (t < n) d[t] = (_Float16)s[t];
}

// Bt[c*R + r] = W[(r+rowOff)*C + c]  (transpose R rows x C cols slice to f16)
__global__ void k_w16T(const float* __restrict__ W, _Float16* __restrict__ Bt,
                       int R, int C, int rowOff) {
  size_t t = (size_t)blockIdx.x * blockDim.x + threadIdx.x;
  if (t >= (size_t)R * C) return;
  int r = (int)(t % R), c = (int)(t / R);
  Bt[t] = (_Float16)W[(size_t)(r + rowOff) * C + c];
}

// out[m, a*D+d] = emb[a, idx[m,a], d]  (embedding gather -> f16)
__global__ void k_embed(const int* __restrict__ idx, const float* __restrict__ emb,
                        _Float16* __restrict__ out, int M, int A, int V, int D) {
  size_t t = (size_t)blockIdx.x * blockDim.x + threadIdx.x;
  if (t >= (size_t)M * A * D) return;
  int d = (int)(t % D);
  int a = (int)((t / D) % A);
  int m = (int)(t / ((size_t)D * A));
  out[t] = (_Float16)emb[((size_t)a * V + idx[(size_t)m * A + a]) * D + d];
}

// out[row] = dot(A[row,:], v) (+ bias1[0]) ; one wave per row
__global__ __launch_bounds__(256)
void k_rowdot(const float* __restrict__ A, const float* __restrict__ v,
              float* __restrict__ out, int M, int C, const float* __restrict__ bias1) {
  int lane = threadIdx.x & 31;
  int row = blockIdx.x * (blockDim.x >> 5) + (threadIdx.x >> 5);
  if (row >= M) return;
  float s = 0.f;
  for (int c = lane; c < C; c += 32) s += A[(size_t)row * C + c] * v[c];
  for (int o = 16; o > 0; o >>= 1) s += __shfl_down(s, o, 32);
  if (lane == 0) out[row] = s + (bias1 ? bias1[0] : 0.f);
}

// GATEConv edge attention: a_e = leaky( sum_c leaky(p[src]+q[e])*att_l + hr[dst] )
__global__ __launch_bounds__(256)
void k_gate_att(const float* __restrict__ p, const float* __restrict__ q,
                const int* __restrict__ ei, const float* __restrict__ attl,
                const float* __restrict__ hr, float* __restrict__ a_e, int E, int C) {
  int lane = threadIdx.x & 31;
  int e = blockIdx.x * (blockDim.x >> 5) + (threadIdx.x >> 5);
  if (e >= E) return;
  int j = ei[e];        // source
  int i = ei[E + e];    // target
  float s = 0.f;
  for (int c = lane; c < C; c += 32) {
    float v = p[(size_t)j * C + c] + q[(size_t)e * C + c];
    s += leakyf(v) * attl[c];
  }
  for (int o = 16; o > 0; o >>= 1) s += __shfl_down(s, o, 32);
  if (lane == 0) a_e[e] = leakyf(s + hr[i]);
}

// GATConv edge attention: a_e = leaky(asrc[src] + adst[dst])
__global__ void k_edge_att(const float* __restrict__ asrc, const float* __restrict__ adst,
                           const int* __restrict__ ei, float* __restrict__ a_e, int E) {
  int e = blockIdx.x * blockDim.x + threadIdx.x;
  if (e >= E) return;
  a_e[e] = leakyf(asrc[ei[e]] + adst[ei[E + e]]);
}

// mol attention: a_n = leaky(hsdot[n] + oddot[batch[n]])
__global__ void k_gather_att(const float* __restrict__ sa, const float* __restrict__ sb,
                             const int* __restrict__ batch, float* __restrict__ a_n, int N) {
  int n = blockIdx.x * blockDim.x + threadIdx.x;
  if (n >= N) return;
  a_n[n] = leakyf(sa[n] + sb[batch[n]]);
}

// softmax pass 1: a <- exp(a); denom[idx] += a
__global__ void k_expsum(float* __restrict__ a, const int* __restrict__ idx,
                         float* __restrict__ denom, int M) {
  int t = blockIdx.x * blockDim.x + threadIdx.x;
  if (t >= M) return;
  float v = __expf(a[t]);
  a[t] = v;
  atomicAdd(&denom[idx[t]], v);
}

// m[dst,c] += (a_e/denom[dst]) * src[srcnode,c]
__global__ void k_agg_edge(const float* __restrict__ src, const int* __restrict__ ei,
                           const float* __restrict__ a_e, const float* __restrict__ denom,
                           float* __restrict__ m, int E, int C) {
  size_t t = (size_t)blockIdx.x * blockDim.x + threadIdx.x;
  if (t >= (size_t)E * C) return;
  int e = (int)(t / C), c = (int)(t % C);
  int j = ei[e], i = ei[E + e];
  float w = a_e[e] / (denom[i] + 1e-16f);
  atomicAdd(&m[(size_t)i * C + c], w * src[(size_t)j * C + c]);
}

// m[batch[n],c] += (a_n/denom[g]) * hs[n,c]
__global__ void k_agg_node(const float* __restrict__ hs, const int* __restrict__ batch,
                           const float* __restrict__ a_n, const float* __restrict__ denom,
                           float* __restrict__ m, int N, int C) {
  size_t t = (size_t)blockIdx.x * blockDim.x + threadIdx.x;
  if (t >= (size_t)N * C) return;
  int n = (int)(t / C), c = (int)(t % C);
  int g = batch[n];
  float w = a_n[n] / (denom[g] + 1e-16f);
  atomicAdd(&m[(size_t)g * C + c], w * hs[t]);
}

// out[batch[n],c] += h[n,c]
__global__ void k_segsum(const float* __restrict__ h, const int* __restrict__ batch,
                         float* __restrict__ out, int N, int C) {
  size_t t = (size_t)blockIdx.x * blockDim.x + threadIdx.x;
  if (t >= (size_t)N * C) return;
  int n = (int)(t / C), c = (int)(t % C);
  atomicAdd(&out[(size_t)batch[n] * C + c], h[t]);
}

__global__ void k_relu16(float* __restrict__ x, _Float16* __restrict__ x16, size_t n) {
  size_t t = (size_t)blockIdx.x * blockDim.x + threadIdx.x;
  if (t >= n) return;
  float v = fmaxf(x[t], 0.f);
  x[t] = v; x16[t] = (_Float16)v;
}

// x <- elu(x + bias[c]); also write f16 mirror
__global__ void k_elu_bias(float* __restrict__ x, const float* __restrict__ bias,
                           _Float16* __restrict__ x16, int M, int C) {
  size_t t = (size_t)blockIdx.x * blockDim.x + threadIdx.x;
  if (t >= (size_t)M * C) return;
  int c = (int)(t % C);
  float v = x[t] + bias[c];
  v = v > 0.f ? v : expm1f(v);
  x[t] = v; x16[t] = (_Float16)v;
}

// GRUCell gate math + relu; writes f32 h and f16 mirror. C fixed = 256, 3C = 768.
__global__ void k_gru(const float* __restrict__ gi, const float* __restrict__ gh,
                      const float* __restrict__ bih, const float* __restrict__ bhh,
                      float* __restrict__ h, _Float16* __restrict__ h16, int M) {
  size_t t = (size_t)blockIdx.x * blockDim.x + threadIdx.x;
  if (t >= (size_t)M * 256) return;
  int n = (int)(t >> 8), c = (int)(t & 255);
  const float* gin = gi + (size_t)n * 768;
  const float* ghn = gh + (size_t)n * 768;
  float ir  = gin[c]       + bih[c];
  float iz  = gin[c + 256] + bih[c + 256];
  float inn = gin[c + 512] + bih[c + 512];
  float hrv = ghn[c]       + bhh[c];
  float hz  = ghn[c + 256] + bhh[c + 256];
  float hn  = ghn[c + 512] + bhh[c + 512];
  float r = 1.f / (1.f + __expf(-(ir + hrv)));
  float z = 1.f / (1.f + __expf(-(iz + hz)));
  float nn = tanhf(inn + r * hn);
  float o = (1.f - z) * nn + z * h[t];
  o = fmaxf(o, 0.f);
  h[t] = o; h16[t] = (_Float16)o;
}

// ---------------------------------------------------------------------------
extern "C" void kernel_launch(void* const* d_in, const int* in_sizes, int n_in,
                              void* d_out, int out_size, void* d_ws, size_t ws_size,
                              hipStream_t stream) {
  const int* x      = (const int*)d_in[0];
  const int* ei     = (const int*)d_in[1];
  const int* ea     = (const int*)d_in[2];
  const int* batch  = (const int*)d_in[3];
  const float* atom_emb = (const float*)d_in[5];
  const float* bond_emb = (const float*)d_in[6];
  const float* lin1_W   = (const float*)d_in[7];
  const float* lin1_b   = (const float*)d_in[8];
  const float* gate_lin1_W = (const float*)d_in[9];
  const float* gate_att_l  = (const float*)d_in[10];
  const float* gate_att_r  = (const float*)d_in[11];
  const float* gate_lin2_W = (const float*)d_in[12];
  const float* gate_bias   = (const float*)d_in[13];
  const float* gru_Wih = (const float*)d_in[14];
  const float* gru_Whh = (const float*)d_in[15];
  const float* gru_bih = (const float*)d_in[16];
  const float* gru_bhh = (const float*)d_in[17];
  const float* conv_W       = (const float*)d_in[18];
  const float* conv_att_src = (const float*)d_in[19];
  const float* conv_att_dst = (const float*)d_in[20];
  const float* conv_bias    = (const float*)d_in[21];
  const float* mol_W       = (const float*)d_in[22];
  const float* mol_att_src = (const float*)d_in[23];
  const float* mol_att_dst = (const float*)d_in[24];
  const float* mol_bias    = (const float*)d_in[25];
  const float* molgru_Wih = (const float*)d_in[26];
  const float* molgru_Whh = (const float*)d_in[27];
  const float* molgru_bih = (const float*)d_in[28];
  const float* molgru_bhh = (const float*)d_in[29];
  const float* lin2_W = (const float*)d_in[30];
  const float* lin2_b = (const float*)d_in[31];
  float* outp = (float*)d_out;

  const int N = in_sizes[0] / 9;     // 80000
  const int E = in_sizes[1] / 2;     // 160000
  const int G = out_size;            // 4096
  const int H = 256, H3 = 768, AD = 288, ED = 96;

  // ---- workspace bump allocator (256B aligned) ----
  size_t off = 0;
  auto alloc = [&](size_t bytes) -> void* {
    void* p = (char*)d_ws + off;
    off += (bytes + 255) & ~(size_t)255;
    return p;
  };
  // weights (f16)
  _Float16* lin1Wt  = (_Float16*)alloc(sizeof(_Float16) * H * AD);       // [256,288]
  _Float16* gW1a    = (_Float16*)alloc(sizeof(_Float16) * H * H);        // [256,256]
  _Float16* gW1b    = (_Float16*)alloc(sizeof(_Float16) * H * ED);       // [256,96]
  _Float16* gW2t    = (_Float16*)alloc(sizeof(_Float16) * H * H);
  _Float16* wih16[3], *whh16[3];
  for (int l = 0; l < 3; ++l) { wih16[l] = (_Float16*)alloc(sizeof(_Float16) * H3 * H);
                                whh16[l] = (_Float16*)alloc(sizeof(_Float16) * H3 * H); }
  _Float16* convWt[2];
  for (int l = 0; l < 2; ++l) convWt[l] = (_Float16*)alloc(sizeof(_Float16) * H * H);
  _Float16* molWt   = (_Float16*)alloc(sizeof(_Float16) * H * H);
  _Float16* mgWih16 = (_Float16*)alloc(sizeof(_Float16) * H3 * H);
  _Float16* mgWhh16 = (_Float16*)alloc(sizeof(_Float16) * H3 * H);
  // activations
  _Float16* xe16  = (_Float16*)alloc(sizeof(_Float16) * (size_t)N * AD);
  _Float16* ee16  = (_Float16*)alloc(sizeof(_Float16) * (size_t)E * ED);
  _Float16* h16   = (_Float16*)alloc(sizeof(_Float16) * (size_t)N * H);
  _Float16* g16   = (_Float16*)alloc(sizeof(_Float16) * (size_t)N * H);
  _Float16* out16 = (_Float16*)alloc(sizeof(_Float16) * (size_t)G * H);
  float* hf    = (float*)alloc(sizeof(float) * (size_t)N * H);
  float* gi    = (float*)alloc(sizeof(float) * (size_t)N * H3);   // also q[E,256]
  float* gh    = (float*)alloc(sizeof(float) * (size_t)N * H3);
  float* hs    = (float*)alloc(sizeof(float) * (size_t)N * H);    // p / hl / conv hs / mol hs
  float* mbuf  = (float*)alloc(sizeof(float) * (size_t)N * H);
  float* a_e   = (float*)alloc(sizeof(float) * (size_t)E);
  float* denom = (float*)alloc(sizeof(float) * (size_t)N);
  float* sc1   = (float*)alloc(sizeof(float) * (size_t)N);
  float* sc2   = (float*)alloc(sizeof(float) * (size_t)N);
  float* outf  = (float*)alloc(sizeof(float) * (size_t)G * H);
  float* od    = (float*)alloc(sizeof(float) * (size_t)G * H);
  if (off > ws_size) return;  // workspace too small; bail deterministically

  const int TB = 256;
  auto nb = [](size_t n, int b) { return (unsigned)((n + b - 1) / b); };
  auto gemm = [&](const _Float16* A, const _Float16* Bt, const float* bias,
                  float* C, _Float16* C16, int M, int Nc, int K, int act) {
    dim3 grid((M + 127) / 128, Nc / 64);
    k_gemm_wmma<<<grid, 128, 0, stream>>>(A, Bt, bias, C, C16, M, Nc, K, act);
  };

  // ---- phase 0: weight conversion ----
  k_w16T<<<nb((size_t)AD * H, TB), TB, 0, stream>>>(lin1_W, lin1Wt, AD, H, 0);
  k_w16T<<<nb((size_t)H * H, TB), TB, 0, stream>>>(gate_lin1_W, gW1a, H, H, 0);
  k_w16T<<<nb((size_t)ED * H, TB), TB, 0, stream>>>(gate_lin1_W, gW1b, ED, H, 256);
  k_w16T<<<nb((size_t)H * H, TB), TB, 0, stream>>>(gate_lin2_W, gW2t, H, H, 0);
  for (int l = 0; l < 3; ++l) {
    k_cast16<<<nb((size_t)H3 * H, TB), TB, 0, stream>>>(gru_Wih + (size_t)l * H3 * H, wih16[l], (size_t)H3 * H);
    k_cast16<<<nb((size_t)H3 * H, TB), TB, 0, stream>>>(gru_Whh + (size_t)l * H3 * H, whh16[l], (size_t)H3 * H);
  }
  for (int l = 0; l < 2; ++l)
    k_w16T<<<nb((size_t)H * H, TB), TB, 0, stream>>>(conv_W + (size_t)l * H * H, convWt[l], H, H, 0);
  k_w16T<<<nb((size_t)H * H, TB), TB, 0, stream>>>(mol_W, molWt, H, H, 0);
  k_cast16<<<nb((size_t)H3 * H, TB), TB, 0, stream>>>(molgru_Wih, mgWih16, (size_t)H3 * H);
  k_cast16<<<nb((size_t)H3 * H, TB), TB, 0, stream>>>(molgru_Whh, mgWhh16, (size_t)H3 * H);

  // ---- phase 1: embeddings ----
  k_embed<<<nb((size_t)N * AD, TB), TB, 0, stream>>>(x, atom_emb, xe16, N, 9, 64, 32);
  k_embed<<<nb((size_t)E * ED, TB), TB, 0, stream>>>(ea, bond_emb, ee16, E, 3, 16, 32);

  // ---- phase 2: h = leaky(xe @ lin1_W + b) ----
  gemm(xe16, lin1Wt, lin1_b, hf, h16, N, H, AD, 1);

  // ---- phase 3: GATEConv + GRU[0] ----
  gemm(h16, gW1a, nullptr, hs, nullptr, N, H, H, 0);             // p = h @ W1[:256]
  gemm(ee16, gW1b, nullptr, gi, nullptr, E, H, ED, 0);           // q = ee @ W1[256:]
  k_rowdot<<<nb((size_t)N, 8), 256, 0, stream>>>(hf, gate_att_r, sc1, N, H, nullptr);
  k_gate_att<<<nb((size_t)E, 8), 256, 0, stream>>>(hs, gi, ei, gate_att_l, sc1, a_e, E, H);
  k_zerof<<<nb((size_t)N, TB), TB, 0, stream>>>(denom, (size_t)N);
  k_expsum<<<nb((size_t)E, TB), TB, 0, stream>>>(a_e, ei + E, denom, E);
  gemm(h16, gW2t, nullptr, hs, nullptr, N, H, H, 0);             // hl = h @ gate_lin2
  k_zerof<<<nb((size_t)N * H, TB), TB, 0, stream>>>(mbuf, (size_t)N * H);
  k_agg_edge<<<nb((size_t)E * H, TB), TB, 0, stream>>>(hs, ei, a_e, denom, mbuf, E, H);
  k_elu_bias<<<nb((size_t)N * H, TB), TB, 0, stream>>>(mbuf, gate_bias, g16, N, H);
  gemm(g16, wih16[0], nullptr, gi, nullptr, N, H3, H, 0);
  gemm(h16, whh16[0], nullptr, gh, nullptr, N, H3, H, 0);
  k_gru<<<nb((size_t)N * H, TB), TB, 0, stream>>>(gi, gh, gru_bih, gru_bhh, hf, h16, N);

  // ---- phase 4: GATConv layers + GRU[1..2] ----
  for (int l = 0; l < 2; ++l) {
    gemm(h16, convWt[l], nullptr, hs, nullptr, N, H, H, 0);
    k_rowdot<<<nb((size_t)N, 8), 256, 0, stream>>>(hs, conv_att_src + l * H, sc1, N, H, nullptr);
    k_rowdot<<<nb((size_t)N, 8), 256, 0, stream>>>(hs, conv_att_dst + l * H, sc2, N, H, nullptr);
    k_edge_att<<<nb((size_t)E, TB), TB, 0, stream>>>(sc1, sc2, ei, a_e, E);
    k_zerof<<<nb((size_t)N, TB), TB, 0, stream>>>(denom, (size_t)N);
    k_expsum<<<nb((size_t)E, TB), TB, 0, stream>>>(a_e, ei + E, denom, E);
    k_zerof<<<nb((size_t)N * H, TB), TB, 0, stream>>>(mbuf, (size_t)N * H);
    k_agg_edge<<<nb((size_t)E * H, TB), TB, 0, stream>>>(hs, ei, a_e, denom, mbuf, E, H);
    k_elu_bias<<<nb((size_t)N * H, TB), TB, 0, stream>>>(mbuf, conv_bias + l * H, g16, N, H);
    gemm(g16, wih16[l + 1], nullptr, gi, nullptr, N, H3, H, 0);
    gemm(h16, whh16[l + 1], nullptr, gh, nullptr, N, H3, H, 0);
    k_gru<<<nb((size_t)N * H, TB), TB, 0, stream>>>(gi, gh, gru_bih + (size_t)(l + 1) * H3,
                                                    gru_bhh + (size_t)(l + 1) * H3, hf, h16, N);
  }

  // ---- phase 5: attentive readout ----
  k_zerof<<<nb((size_t)G * H, TB), TB, 0, stream>>>(outf, (size_t)G * H);
  k_segsum<<<nb((size_t)N * H, TB), TB, 0, stream>>>(hf, batch, outf, N, H);
  k_relu16<<<nb((size_t)G * H, TB), TB, 0, stream>>>(outf, out16, (size_t)G * H);
  for (int t = 0; t < 2; ++t) {
    gemm(h16, molWt, nullptr, hs, nullptr, N, H, H, 0);          // hs = h @ mol_W
    gemm(out16, molWt, nullptr, od, nullptr, G, H, H, 0);        // od = out @ mol_W
    k_rowdot<<<nb((size_t)N, 8), 256, 0, stream>>>(hs, mol_att_src, sc1, N, H, nullptr);
    k_rowdot<<<nb((size_t)G, 8), 256, 0, stream>>>(od, mol_att_dst, sc2, G, H, nullptr);
    k_gather_att<<<nb((size_t)N, TB), TB, 0, stream>>>(sc1, sc2, batch, a_e, N);
    k_zerof<<<nb((size_t)G, TB), TB, 0, stream>>>(denom, (size_t)G);
    k_expsum<<<nb((size_t)N, TB), TB, 0, stream>>>(a_e, batch, denom, N);
    k_zerof<<<nb((size_t)G * H, TB), TB, 0, stream>>>(mbuf, (size_t)G * H);
    k_agg_node<<<nb((size_t)N * H, TB), TB, 0, stream>>>(hs, batch, a_e, denom, mbuf, N, H);
    k_elu_bias<<<nb((size_t)G * H, TB), TB, 0, stream>>>(mbuf, mol_bias, g16, G, H);
    gemm(g16, mgWih16, nullptr, gi, nullptr, G, H3, H, 0);
    gemm(out16, mgWhh16, nullptr, gh, nullptr, G, H3, H, 0);
    k_gru<<<nb((size_t)G * H, TB), TB, 0, stream>>>(gi, gh, molgru_bih, molgru_bhh, outf, out16, G);
  }

  // ---- final: d_out[g] = out[g,:] @ lin2_W + lin2_b ----
  k_rowdot<<<nb((size_t)G, 8), 256, 0, stream>>>(outf, lin2_W, outp, G, H, lin2_b);
}
